// SLSTM_17128329576615
// MI455X (gfx1250) — compile-verified, matrix operand order checked by reading, not run on hardware
//
#include <hip/hip_runtime.h>
#include <hip/hip_bf16.h>

// ---------------------------------------------------------------------------
// sLSTM (T=256, B=512, F=256, H=512, P=12) for MI455X (gfx1250, wave32).
// Per step: Z[512,2048] = [x_t | h_slot] @ Wcat + bias via
// v_wmma_f32_16x16x32_bf16 (f32 accum), then softmax(f,i,o)/tanh(g) + update.
// Weights pre-transposed to N-major bf16 once per call (L2-resident, 3 MB);
// A-tile conversion f32->bf16 fused into the GEMM loader (no pack kernel).
// ---------------------------------------------------------------------------

#define TT 256
#define BB 512
#define FF 256
#define HH 512
#define PP 12
#define KK (FF + HH)        // 768
#define NN (4 * HH)         // 2048
#define KC 32               // K chunk per WMMA stage (divides FF -> clean split)
#define NK (KK / KC)        // 24
#define BM 64               // block tile M
#define BN 128              // block tile N

typedef __attribute__((ext_vector_type(16))) __bf16 v16bf;
typedef __attribute__((ext_vector_type(8)))  __bf16 v8bf;
typedef __attribute__((ext_vector_type(8)))  float  v8f;
typedef __attribute__((ext_vector_type(4)))  float  v4f;

typedef union { v16bf v; v8bf h[2]; } bf16x16;

// ---------------------------------------------------------------- prep ----
// Wt is stored N-major: Wt[n * KK + k], n in [0,2048) = gate*512 + col.
__global__ __launch_bounds__(256) void conv_weights_kernel(
    const float* __restrict__ Wf, const float* __restrict__ Wi,
    const float* __restrict__ Wo, const float* __restrict__ Wg,
    __bf16* __restrict__ Wt) {
  int i = blockIdx.x * 256 + threadIdx.x;          // over NN*KK
  if (i >= NN * KK) return;
  int n = i / KK, k = i % KK;
  int gate = n >> 9, col = n & 511;
  const float* W = (gate == 0) ? Wf : (gate == 1) ? Wi : (gate == 2) ? Wo : Wg;
  Wt[i] = (__bf16)W[k * HH + col];
}

__global__ __launch_bounds__(256) void conv_bias_kernel(
    const float* __restrict__ bf_, const float* __restrict__ bi_,
    const float* __restrict__ bo_, const float* __restrict__ bg_,
    float* __restrict__ bias) {
  int n = blockIdx.x * 256 + threadIdx.x;
  if (n >= NN) return;
  int gate = n >> 9, col = n & 511;
  const float* b = (gate == 0) ? bf_ : (gate == 1) ? bi_ : (gate == 2) ? bo_ : bg_;
  bias[n] = b[col];
}

__global__ __launch_bounds__(256) void zero_f32_kernel(float* __restrict__ p, int n) {
  int i = blockIdx.x * 256 + threadIdx.x;
  if (i < n) p[i] = 0.0f;
}

// ---------------------------------------------------------------- GEMM ----
// Z[512,2048] = [x_t | h_slot](bf16) @ Wt^T + bias.
// 128 threads = 4 waves (2x2), wave tile 32x64, block tile 64x128.
__global__ __launch_bounds__(128) void gemm_wmma_kernel(
    const float* __restrict__ xt, const float* __restrict__ hslot,
    const __bf16* __restrict__ Wt, const float* __restrict__ bias,
    float* __restrict__ Z) {
  __shared__ __bf16 sA[2][BM][KC];     // [m][k] : 8 KB
  __shared__ __bf16 sB[2][BN][KC];     // [n][k] : 16 KB (pre-transposed source)

  const int tid  = threadIdx.x;
  const int wave = tid >> 5;
  const int lane = tid & 31;
  const int wm = (wave >> 1) * 32;                 // wave M offset in tile
  const int wn = (wave & 1) * 64;                  // wave N offset in tile
  const int bm = blockIdx.x * BM;
  const int bn = blockIdx.y * BN;

  v8f acc[2][4];
#pragma unroll
  for (int mi = 0; mi < 2; ++mi)
#pragma unroll
    for (int ni = 0; ni < 4; ++ni) acc[mi][ni] = (v8f)0.0f;

  const int khalf = (lane >> 4) * 8;               // 0 | 8 (16-bit A/B layout)

  auto load_stage = [&](int kc, int buf) {
    const int k0 = kc * KC;
    // A tile: 64x32, f32 source (x or h, chunk never straddles), cvt to bf16.
    const bool fromX = (k0 < FF);
    const float* src = fromX ? xt : hslot;
    const int sstride = fromX ? FF : HH;
    const int kb = fromX ? k0 : (k0 - FF);
#pragma unroll
    for (int it = 0; it < 2; ++it) {
      int r = (tid >> 2) + it * 32;
      int kofs = (tid & 3) * 8;
      const float* p = &src[(size_t)(bm + r) * sstride + kb + kofs];
      v4f f0 = *(const v4f*)p;
      v4f f1 = *(const v4f*)(p + 4);
      v8bf v;
      v[0] = (__bf16)f0[0]; v[1] = (__bf16)f0[1];
      v[2] = (__bf16)f0[2]; v[3] = (__bf16)f0[3];
      v[4] = (__bf16)f1[0]; v[5] = (__bf16)f1[1];
      v[6] = (__bf16)f1[2]; v[7] = (__bf16)f1[3];
      *(v8bf*)&sA[buf][r][kofs] = v;
    }
    // B tile: 128x32 from N-major Wt -> contiguous b128 load + b128 store.
#pragma unroll
    for (int it = 0; it < 4; ++it) {
      int n = (tid >> 2) + it * 32;
      int kofs = (tid & 3) * 8;
      v8bf v = *(const v8bf*)&Wt[(size_t)(bn + n) * KK + k0 + kofs];
      *(v8bf*)&sB[buf][n][kofs] = v;
    }
  };

  load_stage(0, 0);
  __syncthreads();

  for (int kc = 0; kc < NK; ++kc) {
    const int cur = kc & 1;
    if (kc + 1 < NK) load_stage(kc + 1, cur ^ 1);

    bf16x16 bfrag[4];
#pragma unroll
    for (int ni = 0; ni < 4; ++ni) {
      int bcol = wn + ni * 16 + (lane & 15);
      bfrag[ni].h[0] = *(const v8bf*)&sB[cur][bcol][khalf];
      bfrag[ni].h[1] = *(const v8bf*)&sB[cur][bcol][khalf + 16];
    }
#pragma unroll
    for (int mi = 0; mi < 2; ++mi) {
      int arow = wm + mi * 16 + (lane & 15);
      bf16x16 afrag;
      afrag.h[0] = *(const v8bf*)&sA[cur][arow][khalf];
      afrag.h[1] = *(const v8bf*)&sA[cur][arow][khalf + 16];
#pragma unroll
      for (int ni = 0; ni < 4; ++ni) {
        acc[mi][ni] = __builtin_amdgcn_wmma_f32_16x16x32_bf16(
            false, afrag.v, false, bfrag[ni].v,
            (short)0, acc[mi][ni], false, false);
      }
    }
    __syncthreads();
  }

  // Epilogue: C/D layout — VGPR r: M = r + (lane>=16 ? 8 : 0), N = lane%16
  const int mtop = (lane >> 4) * 8;
#pragma unroll
  for (int mi = 0; mi < 2; ++mi) {
#pragma unroll
    for (int ni = 0; ni < 4; ++ni) {
      int col = bn + wn + ni * 16 + (lane & 15);
      float bv = bias[col];
      int rowbase = bm + wm + mi * 16 + mtop;
#pragma unroll
      for (int r = 0; r < 8; ++r)
        Z[(size_t)(rowbase + r) * NN + col] = acc[mi][ni][r] + bv;
    }
  }
}

// ---------------------------------------------------- softmax + update ----
__device__ __forceinline__ float block_reduce(float v, bool is_max, float* sm) {
#pragma unroll
  for (int o = 16; o > 0; o >>= 1) {
    float t = __shfl_xor(v, o);
    v = is_max ? fmaxf(v, t) : (v + t);
  }
  int lane = threadIdx.x & 31, w = threadIdx.x >> 5;
  if (lane == 0) sm[w] = v;
  __syncthreads();
  float r = sm[0];
#pragma unroll
  for (int i = 1; i < 8; ++i) r = is_max ? fmaxf(r, sm[i]) : (r + sm[i]);
  __syncthreads();
  return r;
}

// One block (256 threads) per batch row; each thread handles h and h+256.
__global__ __launch_bounds__(256) void act_kernel(
    const float* __restrict__ Z, const float* __restrict__ c_prev,
    float* __restrict__ c_cur, float* __restrict__ h_slot, int idx) {
  __shared__ float sm[8];
  const int b = blockIdx.x;
  const int h0 = threadIdx.x, h1 = threadIdx.x + 256;
  const float* zrow = Z + (size_t)b * NN;

  float gate0[3], gate1[3];
#pragma unroll
  for (int g = 0; g < 3; ++g) {                    // f, i, o softmax over H
    float z0 = zrow[g * HH + h0];
    float z1 = zrow[g * HH + h1];
    float m = block_reduce(fmaxf(z0, z1), true, sm);
    float e0 = __expf(z0 - m), e1 = __expf(z1 - m);
    float s = block_reduce(e0 + e1, false, sm);
    float inv = 1.0f / s;
    gate0[g] = e0 * inv;
    gate1[g] = e1 * inv;
  }
  float g0 = tanhf(zrow[3 * HH + h0]);
  float g1 = tanhf(zrow[3 * HH + h1]);

  // c_prev[idx,:] is a single row broadcast over the batch (reference quirk)
  float cp0 = c_prev[(size_t)idx * HH + h0];
  float cp1 = c_prev[(size_t)idx * HH + h1];
  float cn0 = tanhf(gate0[0] * cp0 + gate0[1] * g0);
  float cn1 = tanhf(gate1[0] * cp1 + gate1[1] * g1);

  c_cur[(size_t)b * HH + h0] = cn0;
  c_cur[(size_t)b * HH + h1] = cn1;
  h_slot[(size_t)b * HH + h0] = gate0[2] * cn0;
  h_slot[(size_t)b * HH + h1] = gate1[2] * cn1;
}

__global__ __launch_bounds__(256) void copy_out_kernel(
    const float* __restrict__ src, float* __restrict__ dst, int n) {
  int i = blockIdx.x * 256 + threadIdx.x;
  if (i < n) dst[i] = src[i];
}

// ---------------------------------------------------------------- host ----
extern "C" void kernel_launch(void* const* d_in, const int* in_sizes, int n_in,
                              void* d_out, int out_size, void* d_ws, size_t ws_size,
                              hipStream_t stream) {
  const float* x   = (const float*)d_in[0];
  const float* Wf  = (const float*)d_in[1];
  const float* bf_ = (const float*)d_in[2];
  const float* Wi  = (const float*)d_in[3];
  const float* bi_ = (const float*)d_in[4];
  const float* Wo  = (const float*)d_in[5];
  const float* bo_ = (const float*)d_in[6];
  const float* Wg  = (const float*)d_in[7];
  const float* bg_ = (const float*)d_in[8];

  char* ws = (char*)d_ws;
  size_t off = 0;
  auto take = [&](size_t bytes) {
    size_t o = off;
    off += (bytes + 255) & ~(size_t)255;
    return o;
  };
  __bf16* Wt      = (__bf16*)(ws + take((size_t)NN * KK * sizeof(__bf16)));    // 3 MB
  float*  biasAll = (float*) (ws + take((size_t)NN * sizeof(float)));
  float*  Zbuf    = (float*) (ws + take((size_t)BB * NN * sizeof(float)));     // 4 MB
  float*  h_slots = (float*) (ws + take((size_t)PP * BB * HH * sizeof(float))); // 12 MB
  float*  cbuf    = (float*) (ws + take((size_t)2 * BB * HH * sizeof(float)));  // 2 MB

  // One-time (per call) prep: weight transpose+cvt, bias pack, state zeroing.
  conv_weights_kernel<<<(NN * KK + 255) / 256, 256, 0, stream>>>(Wf, Wi, Wo, Wg, Wt);
  conv_bias_kernel<<<(NN + 255) / 256, 256, 0, stream>>>(bf_, bi_, bo_, bg_, biasAll);
  zero_f32_kernel<<<(PP * BB * HH + 255) / 256, 256, 0, stream>>>(h_slots, PP * BB * HH);
  zero_f32_kernel<<<(2 * BB * HH + 255) / 256, 256, 0, stream>>>(cbuf, 2 * BB * HH);

  dim3 ggrid(BB / BM, NN / BN);  // 8 x 16 = 128 workgroups
  for (int t = 0; t < TT; ++t) {
    int idx = t % PP;
    gemm_wmma_kernel<<<ggrid, 128, 0, stream>>>(
        x + (size_t)t * BB * FF, h_slots + (size_t)idx * BB * HH,
        Wt, biasAll, Zbuf);
    act_kernel<<<BB, 256, 0, stream>>>(
        Zbuf, cbuf + (size_t)(t & 1) * BB * HH,
        cbuf + (size_t)((t + 1) & 1) * BB * HH,
        h_slots + (size_t)idx * BB * HH, idx);
  }

  int fin = (TT - 1) % PP;  // = 3
  copy_out_kernel<<<(BB * HH + 255) / 256, 256, 0, stream>>>(
      h_slots + (size_t)fin * BB * HH, (float*)d_out, BB * HH);
}
